// KVCacheHeavyHitters_49563922596461
// MI455X (gfx1250) — compile-verified, matrix operand order checked by reading, not run on hardware
//
#include <hip/hip_runtime.h>
#include <hip/hip_bf16.h>
#include <stdint.h>

// Problem dims (fixed by reference): B=1, H=32, L=4096, D=128, W=400
#define H_ 32
#define L_ 4096
#define D_ 128
#define W_ 400

typedef __attribute__((ext_vector_type(8))) int v8i;

// ---------------------------------------------------------------------------
// Kernel 0: init per-head argmax keys (must run every call: ws is not re-zeroed)
// ---------------------------------------------------------------------------
__global__ void hh_init(unsigned long long* __restrict__ best) {
    if (threadIdx.x < H_) best[threadIdx.x] = 0ull;
}

// ---------------------------------------------------------------------------
// Kernel 1: negative-count per slot via V_WMMA_I32_16X16X64_IU8, then packed
// u64 atomicMax argmax per head.
//
// One wave handles a tile of 16 slots (rows) of one head. Per WMMA step it
// consumes 64 history entries per row: lane l covers 32 contiguous floats of
// row (l&15); the two 32-wide halves of the K=64 chunk are split across lanes
// l and l+16. Indicators (f < 0 -> 1) are packed 4-per-dword into the 8
// A-operand VGPRs. B = all 0x01 bytes, so D row sums = counts (the within-row
// K ordering of the 8-bit A layout is irrelevant for an all-ones B).
//
// Tail (W=400 = 6*64 + 16): branch-free. All lanes load the same 4 valid
// float4 groups of their row (lanes l and l+16 share a row pointer, so the
// duplicate loads coalesce); upper-half lanes mask their contribution to 0.
// ---------------------------------------------------------------------------
__global__ __launch_bounds__(256) void hh_count_argmax(
    const float* __restrict__ attn, unsigned long long* __restrict__ best) {
    const int lane     = threadIdx.x & 31;
    const int waveId   = threadIdx.x >> 5;
    const int tile     = blockIdx.x * (blockDim.x >> 5) + waveId;  // [0, H*L/16)
    const int tilesPerHead = L_ / 16;                              // 256
    const int h        = tile / tilesPerHead;
    const int slotBase = (tile - h * tilesPerHead) * 16;
    const int row      = lane & 15;   // matrix row M = lane % 16
    const int half     = lane >> 4;   // which 32-wide half of the K=64 chunk

    const float* __restrict__ p =
        attn + ((size_t)h * L_ + (size_t)(slotBase + row)) * W_;

    v8i ones;
#pragma unroll
    for (int j = 0; j < 8; ++j) ones[j] = 0x01010101;
    v8i acc = {};

    // 6 full K=64 chunks (no bounds checks: max offset 5*64+32+31 = 383 < 400)
    for (int i = 0; i < 6; ++i) {
        const int base = i * 64 + half * 32;
        v8i a;
#pragma unroll
        for (int j = 0; j < 8; ++j) {
            float4 f = *(const float4*)(p + base + j * 4);
            uint32_t b0 = f.x < 0.0f ? 1u : 0u;
            uint32_t b1 = f.y < 0.0f ? 1u : 0u;
            uint32_t b2 = f.z < 0.0f ? 1u : 0u;
            uint32_t b3 = f.w < 0.0f ? 1u : 0u;
            a[j] = (int)(b0 | (b1 << 8) | (b2 << 16) | (b3 << 24));
        }
        acc = __builtin_amdgcn_wmma_i32_16x16x64_iu8(false, a, false, ones,
                                                     acc, false, false);
    }

    // Branch-free tail: entries [384, 400) = 4 float4 groups, counted once
    // (lower-half lanes); upper-half lanes AND with 0. No exec divergence.
    {
        const uint32_t tmask = (half == 0) ? 0xFFFFFFFFu : 0u;
        v8i a;
#pragma unroll
        for (int j = 0; j < 4; ++j) {
            float4 f = *(const float4*)(p + 384 + j * 4);
            uint32_t b0 = f.x < 0.0f ? 1u : 0u;
            uint32_t b1 = f.y < 0.0f ? 1u : 0u;
            uint32_t b2 = f.z < 0.0f ? 1u : 0u;
            uint32_t b3 = f.w < 0.0f ? 1u : 0u;
            a[j] = (int)((b0 | (b1 << 8) | (b2 << 16) | (b3 << 24)) & tmask);
        }
#pragma unroll
        for (int j = 4; j < 8; ++j) a[j] = 0;
        acc = __builtin_amdgcn_wmma_i32_16x16x64_iu8(false, a, false, ones,
                                                     acc, false, false);
    }

    // i32 16x16 C/D layout: lanes 0-15 VGPR j -> M=j ; lanes 16-31 -> M=j+8.
    // Pack (count << 32) | (~slot) so u64 max == (max count, min slot index),
    // matching jnp.argmax first-index tie-break.
    unsigned long long key = 0ull;
#pragma unroll
    for (int j = 0; j < 8; ++j) {
        uint32_t slot = (uint32_t)(slotBase + j + half * 8);
        unsigned long long k =
            ((unsigned long long)(uint32_t)acc[j] << 32) | (0xFFFFFFFFu - slot);
        key = (k > key) ? k : key;
    }
    // Combine the two half-wave row sets (lanes l <-> l+16)
    uint32_t lo  = (uint32_t)key, hi = (uint32_t)(key >> 32);
    uint32_t olo = (uint32_t)__shfl_xor((int)lo, 16, 32);
    uint32_t ohi = (uint32_t)__shfl_xor((int)hi, 16, 32);
    unsigned long long other = ((unsigned long long)ohi << 32) | olo;
    key = (other > key) ? other : key;

    if (lane == 0) atomicMax(&best[h], key);
}

// ---------------------------------------------------------------------------
// Kernel 2: float4 scatter-copy of stacked [2,H,L,D] k/v caches, replacing the
// evicted row of each head with k_val / v_val.
// ---------------------------------------------------------------------------
__global__ __launch_bounds__(256) void hh_kv_scatter(
    const float* __restrict__ k_cache, const float* __restrict__ v_cache,
    const float* __restrict__ k_val, const float* __restrict__ v_val,
    const unsigned long long* __restrict__ best, float* __restrict__ out) {
    const size_t nq = (size_t)2 * H_ * L_ * (D_ / 4);  // 8,388,608 float4s
    for (size_t q = blockIdx.x * (size_t)blockDim.x + threadIdx.x; q < nq;
         q += (size_t)gridDim.x * blockDim.x) {
        const int dq    = (int)(q & (D_ / 4 - 1));   // quad within row
        const size_t r  = q >> 5;                    // row id over [2,H,L]
        const int l     = (int)(r & (L_ - 1));
        const size_t g  = r >> 12;                   // / L
        const int h     = (int)(g & (H_ - 1));
        const int kv    = (int)(g >> 5);             // 0 = k, 1 = v
        const int evict = (int)(0xFFFFFFFFu - (uint32_t)(best[h] & 0xFFFFFFFFu));

        const float* src;
        if (l == evict)
            src = (kv ? v_val : k_val) + (size_t)h * D_ + (size_t)dq * 4;
        else
            src = (kv ? v_cache : k_cache) + ((size_t)h * L_ + l) * D_ +
                  (size_t)dq * 4;
        *(float4*)(out + q * 4) = *(const float4*)src;
    }
}

// ---------------------------------------------------------------------------
// Kernel 3: pos_new / cts_new (int state, emitted numerically as float since
// d_out is a float buffer for this reference).
// ---------------------------------------------------------------------------
__global__ __launch_bounds__(256) void hh_poscts(
    const int* __restrict__ pos, const int* __restrict__ cts,
    const int* __restrict__ input_pos,
    const unsigned long long* __restrict__ best, float* __restrict__ out) {
    const int n   = H_ * L_;
    const int idx = blockIdx.x * blockDim.x + threadIdx.x;
    if (idx >= n) return;
    const int h     = idx >> 12;
    const int l     = idx & (L_ - 1);
    const int evict = (int)(0xFFFFFFFFu - (uint32_t)(best[h] & 0xFFFFFFFFu));
    const int pv = (l == evict) ? input_pos[0] : pos[idx];
    const int cv = (l == evict) ? 0 : cts[idx];
    out[idx]     = (float)pv;
    out[n + idx] = (float)cv;
}

// ---------------------------------------------------------------------------
extern "C" void kernel_launch(void* const* d_in, const int* in_sizes, int n_in,
                              void* d_out, int out_size, void* d_ws,
                              size_t ws_size, hipStream_t stream) {
    const float* attn      = (const float*)d_in[0];  // [1,32,4096,400] f32
    const float* k_cache   = (const float*)d_in[1];  // [1,32,4096,128] f32
    const float* v_cache   = (const float*)d_in[2];  // [1,32,4096,128] f32
    const int*   pos       = (const int*)d_in[3];    // [1,32,4096] i32
    const int*   cts       = (const int*)d_in[4];    // [1,32,4096] i32
    const float* k_val     = (const float*)d_in[5];  // [1,32,1,128] f32
    const float* v_val     = (const float*)d_in[6];  // [1,32,1,128] f32
    const int*   input_pos = (const int*)d_in[7];    // [1] i32

    float* out = (float*)d_out;
    unsigned long long* best = (unsigned long long*)d_ws;  // 32 x u64

    hh_init<<<1, 32, 0, stream>>>(best);

    // H*L/16 = 8192 wave-tiles, 8 waves (256 threads) per block -> 1024 blocks
    hh_count_argmax<<<(H_ * L_ / 16) / 8, 256, 0, stream>>>(attn, best);

    hh_kv_scatter<<<4096, 256, 0, stream>>>(k_cache, v_cache, k_val, v_val,
                                            best, out);

    hh_poscts<<<(H_ * L_) / 256, 256, 0, stream>>>(
        pos, cts, input_pos, best, out + (size_t)2 * H_ * L_ * D_);
}
